// GFAN_33062658245166
// MI455X (gfx1250) — matrix operand order repaired and checked by reading, not compiled, not run on hardware
//
#include <hip/hip_runtime.h>
#include <math.h>

// ---------------------------------------------------------------------------
// GFAN forward for MI455X (gfx1250, wave32). All large GEMMs use
// V_WMMA_F32_16X16X32_BF16 (f32 accumulate, bf16 operands staged via LDS,
// software-pipelined; uniform-split staging and epilogues keep the main
// paths free of exec-mask churn). Workspace: 5*N*512 + 3*N*2048 floats.
// ---------------------------------------------------------------------------

#define B_   64
#define L_   321
#define D_   512
#define DFF_ 2048
#define H_   8
#define NTOK (B_ * L_)   // 20544

typedef __attribute__((ext_vector_type(16))) __bf16 v16bf;
typedef __attribute__((ext_vector_type(4)))  __bf16 v4bf;
typedef __attribute__((ext_vector_type(8)))  float  v8f;

enum { ACT_NONE = 0, ACT_RELU = 1, ACT_LEAKY = 2 };

template <int A>
__device__ __forceinline__ float act_t(float x) {
  if constexpr (A == ACT_RELU)       return x > 0.f ? x : 0.f;
  else if constexpr (A == ACT_LEAKY) return x > 0.f ? x : 0.5f * x;  // alpha=0.5
  else                               return x;
}

template <int PREACT>
__device__ __forceinline__ v4bf cvt4(float4 f) {
  v4bf h;
  h[0] = (__bf16)act_t<PREACT>(f.x);
  h[1] = (__bf16)act_t<PREACT>(f.y);
  h[2] = (__bf16)act_t<PREACT>(f.z);
  h[3] = (__bf16)act_t<PREACT>(f.w);
  return h;
}

__device__ __forceinline__ float wave_max32(float v) {
#pragma unroll
  for (int off = 16; off > 0; off >>= 1) v = fmaxf(v, __shfl_xor(v, off, 32));
  return v;
}
__device__ __forceinline__ float wave_sum32(float v) {
#pragma unroll
  for (int off = 16; off > 0; off >>= 1) v += __shfl_xor(v, off, 32);
  return v;
}

// ---------------------------------------------------------------------------
// WMMA fragment builders (CDNA5 ISA, wave32); 8-element runs contiguous
// in LDS -> ds_load_b128.
// ---------------------------------------------------------------------------
#define LDSPAD 40   // bf16 elems per row (80B stride, conflict-free, 8B-aligned)

__device__ __forceinline__ v16bf frag_a(const __bf16* As, int arow, int lane) {
  v16bf f;
  const __bf16* p = As + arow * LDSPAD + ((lane & 16) ? 8 : 0);
#pragma unroll
  for (int e = 0; e < 8; ++e) { f[e] = p[e]; f[e + 8] = p[e + 16]; }
  return f;
}
__device__ __forceinline__ v16bf frag_b(const __bf16* Bs, int bcol, int lane) {
  v16bf f;
  const __bf16* p = Bs + bcol * LDSPAD + ((lane & 16) ? 8 : 0);
#pragma unroll
  for (int e = 0; e < 8; ++e) { f[e] = p[e]; f[e + 8] = p[e + 16]; }
  return f;
}

// ---------------------------------------------------------------------------
// Generic GEMM: C[M,N] = ACT( A[M,K] @ W[K,N] + bias[N] (+ res[M,N]) ),
// PREACT applied to A while staging. Tile 128x64, BK=32, 256 threads,
// 8 waves x (16 rows x 4 col-tiles), double-buffered LDS, reg-staged pipeline.
// Requires: K % 32 == 0, N % 64 == 0.
// ---------------------------------------------------------------------------
template <int ACT, int PREACT, bool HAS_RES>
__global__ __launch_bounds__(256)
void gemm_ba(const float* __restrict__ A, const float* __restrict__ W,
             const float* __restrict__ bias, const float* __restrict__ res,
             float* __restrict__ C, int M, int K, int N)
{
  __shared__ __bf16 As[2][128 * LDSPAD];
  __shared__ __bf16 Bs[2][64 * LDSPAD];

  const int tid  = threadIdx.x;
  const int lane = tid & 31;
  const int wave = tid >> 5;
  const int mBlk = blockIdx.x * 128;
  const int nBlk = blockIdx.y * 64;
  const int mW   = wave * 16;
  const bool fullM = (mBlk + 128 <= M);   // block-uniform

  auto loadA = [&](int k0, float4* fa) {
    if (fullM) {              // branch-free fast path (160/161 blocks)
#pragma unroll
      for (int t = 0; t < 4; ++t) {
        int idx4 = t * 256 + tid;
        int r = idx4 >> 3, c4 = (idx4 & 7) << 2;
        fa[t] = *(const float4*)(A + (size_t)(mBlk + r) * K + k0 + c4);
      }
    } else {                  // clamped address + cndmask zero (no exec churn)
#pragma unroll
      for (int t = 0; t < 4; ++t) {
        int idx4 = t * 256 + tid;
        int r = idx4 >> 3, c4 = (idx4 & 7) << 2;
        int gr = mBlk + r;
        int grc = gr < M ? gr : M - 1;
        float4 f = *(const float4*)(A + (size_t)grc * K + k0 + c4);
        bool ok = gr < M;
        f.x = ok ? f.x : 0.f; f.y = ok ? f.y : 0.f;
        f.z = ok ? f.z : 0.f; f.w = ok ? f.w : 0.f;
        fa[t] = f;
      }
    }
  };
  auto storeA = [&](__bf16* dst, const float4* fa) {
#pragma unroll
    for (int t = 0; t < 4; ++t) {
      int idx4 = t * 256 + tid;
      int r = idx4 >> 3, c4 = (idx4 & 7) << 2;
      *(v4bf*)(dst + r * LDSPAD + c4) = cvt4<PREACT>(fa[t]);
    }
  };
  auto loadB = [&](int k0, float4* fb) {
#pragma unroll
    for (int t = 0; t < 2; ++t) {
      int idx4 = t * 256 + tid;
      int kk = idx4 >> 4, n4 = (idx4 & 15) << 2;
      fb[t] = *(const float4*)(W + (size_t)(k0 + kk) * N + nBlk + n4);
    }
  };
  auto storeB = [&](__bf16* dst, const float4* fb) {
#pragma unroll
    for (int t = 0; t < 2; ++t) {
      int idx4 = t * 256 + tid;
      int kk = idx4 >> 4, n4 = (idx4 & 15) << 2;
      v4bf h = cvt4<ACT_NONE>(fb[t]);
      dst[(n4 + 0) * LDSPAD + kk] = h[0];   // [n][k] transpose
      dst[(n4 + 1) * LDSPAD + kk] = h[1];
      dst[(n4 + 2) * LDSPAD + kk] = h[2];
      dst[(n4 + 3) * LDSPAD + kk] = h[3];
    }
  };

  v8f acc[4] = {};
  float4 fa[4], fb[2];
  loadA(0, fa); loadB(0, fb);
  storeA(As[0], fa); storeB(Bs[0], fb);
  __syncthreads();

  int cur = 0;
  for (int k0 = 0; k0 < K; k0 += 32) {
    const int nxt = cur ^ 1;
    const bool more = (k0 + 32 < K);
    if (more) { loadA(k0 + 32, fa); loadB(k0 + 32, fb); }  // issue early
    if (k0 + 64 < K) {   // global_prefetch_b8 two tiles ahead
      __builtin_prefetch(A + (size_t)(mBlk + (tid >> 1)) * K + k0 + 64, 0, 1);
      __builtin_prefetch(W + (size_t)(k0 + 64 + (tid >> 3)) * N + nBlk, 0, 1);
    }
    v16bf af = frag_a(As[cur], mW + (lane & 15), lane);
#pragma unroll
    for (int t = 0; t < 4; ++t) {
      v16bf bf = frag_b(Bs[cur], t * 16 + (lane & 15), lane);
      acc[t] = __builtin_amdgcn_wmma_f32_16x16x32_bf16(
          false, af, false, bf, (short)0, acc[t], false, false);
    }
    if (more) { storeA(As[nxt], fa); storeB(Bs[nxt], fb); }  // waits land here
    __syncthreads();
    cur = nxt;
  }

  // epilogue: uniform split; strength-reduced addressing
  const int rowBase = mBlk + mW + ((lane & 16) ? 8 : 0);
  if (fullM) {
#pragma unroll
    for (int t = 0; t < 4; ++t) {
      const int col = nBlk + t * 16 + (lane & 15);
      const float bc = bias[col];
      size_t base = (size_t)rowBase * N + col;
#pragma unroll
      for (int r = 0; r < 8; ++r, base += N) {
        float v = acc[t][r] + bc;
        if constexpr (HAS_RES) v += res[base];
        C[base] = act_t<ACT>(v);
      }
    }
  } else {
#pragma unroll
    for (int t = 0; t < 4; ++t) {
      const int col = nBlk + t * 16 + (lane & 15);
      const float bc = bias[col];
      size_t base = (size_t)rowBase * N + col;
#pragma unroll
      for (int r = 0; r < 8; ++r, base += N) {
        if (rowBase + r < M) {
          float v = acc[t][r] + bc;
          if constexpr (HAS_RES) v += res[base];
          C[base] = act_t<ACT>(v);
        }
      }
    }
  }
}

// ---------------------------------------------------------------------------
// gattn raw scores: scores[b,h,i,j] = sum_e q[b,i,h,e]*k[b,j,h,e]  (E=64)
// ---------------------------------------------------------------------------
__global__ __launch_bounds__(256)
void gattn_scores(const float* __restrict__ q, const float* __restrict__ k,
                  float* __restrict__ scores)
{
  __shared__ __bf16 Qs[128 * LDSPAD];
  __shared__ __bf16 Ks[64 * LDSPAD];   // [j][e]

  const int tid  = threadIdx.x;
  const int lane = tid & 31;
  const int wave = tid >> 5;
  const int mBlk = blockIdx.x * 128;
  const int nBlk = blockIdx.y * 64;
  const int bh   = blockIdx.z;
  const int b    = bh >> 3, h = bh & 7;
  const float* qb = q + (size_t)b * L_ * D_ + h * 64;
  const float* kb = k + (size_t)b * L_ * D_ + h * 64;
  const int mW = wave * 16;

  v8f acc[4] = {};

  for (int k0 = 0; k0 < 64; k0 += 32) {
#pragma unroll
    for (int t = 0; t < 4; ++t) {        // Q tile 128x32
      int idx4 = t * 256 + tid;
      int r = idx4 >> 3, c4 = (idx4 & 7) << 2;
      int gr = mBlk + r;
      int grc = gr < L_ ? gr : L_ - 1;
      float4 f = *(const float4*)(qb + (size_t)grc * D_ + k0 + c4);
      bool ok = gr < L_;
      f.x = ok ? f.x : 0.f; f.y = ok ? f.y : 0.f;
      f.z = ok ? f.z : 0.f; f.w = ok ? f.w : 0.f;
      *(v4bf*)(Qs + r * LDSPAD + c4) = cvt4<ACT_NONE>(f);
    }
#pragma unroll
    for (int t = 0; t < 2; ++t) {        // K tile 64(j)x32(e)
      int idx4 = t * 256 + tid;
      int r = idx4 >> 3, c4 = (idx4 & 7) << 2;
      int gn = nBlk + r;
      int gnc = gn < L_ ? gn : L_ - 1;
      float4 f = *(const float4*)(kb + (size_t)gnc * D_ + k0 + c4);
      bool ok = gn < L_;
      f.x = ok ? f.x : 0.f; f.y = ok ? f.y : 0.f;
      f.z = ok ? f.z : 0.f; f.w = ok ? f.w : 0.f;
      *(v4bf*)(Ks + r * LDSPAD + c4) = cvt4<ACT_NONE>(f);
    }
    __syncthreads();
    v16bf af = frag_a(Qs, mW + (lane & 15), lane);
#pragma unroll
    for (int t = 0; t < 4; ++t) {
      v16bf bf = frag_b(Ks, t * 16 + (lane & 15), lane);
      acc[t] = __builtin_amdgcn_wmma_f32_16x16x32_bf16(
          false, af, false, bf, (short)0, acc[t], false, false);
    }
    __syncthreads();
  }

  float* out = scores + (size_t)bh * L_ * L_;
  const int rowBase = mBlk + mW + ((lane & 16) ? 8 : 0);
  const bool fullT = (mBlk + 128 <= L_) && (nBlk + 64 <= L_);
  if (fullT) {                 // 10/18 tiles: branch-free stores
#pragma unroll
    for (int t = 0; t < 4; ++t) {
      const int col = nBlk + t * 16 + (lane & 15);
      size_t base = (size_t)rowBase * L_ + col;
#pragma unroll
      for (int r = 0; r < 8; ++r, base += L_) out[base] = acc[t][r];
    }
  } else {
#pragma unroll
    for (int t = 0; t < 4; ++t) {
      const int col = nBlk + t * 16 + (lane & 15);
      size_t base = (size_t)rowBase * L_ + col;
#pragma unroll
      for (int r = 0; r < 8; ++r, base += L_)
        if (rowBase + r < L_ && col < L_) out[base] = acc[t][r];
    }
  }
}

// ---------------------------------------------------------------------------
// Top-10 threshold + masked softmax in place; one wave32 per row of 321.
// kth-largest with duplicate counting (iterative max, single-instance removal).
// ---------------------------------------------------------------------------
__global__ __launch_bounds__(256)
void topk_softmax(float* __restrict__ Amat)
{
  const int lane = threadIdx.x & 31;
  const long long row = (long long)blockIdx.x * 8 + (threadIdx.x >> 5);
  float* p = Amat + row * L_;

  float vals[11];
#pragma unroll
  for (int i = 0; i < 11; ++i) {
    int idx = i * 32 + lane;
    vals[i] = (idx < L_) ? p[idx] : -INFINITY;
  }

  unsigned removed = 0;
  float kth = -INFINITY, m0 = -INFINITY;
  for (int it = 0; it < 10; ++it) {
    float lm = -INFINITY; int li = -1;
#pragma unroll
    for (int i = 0; i < 11; ++i)
      if (!((removed >> i) & 1u) && vals[i] > lm) { lm = vals[i]; li = i; }
    float gm = wave_max32(lm);
    if (it == 0) m0 = gm;
    unsigned long long bm = __ballot(lm == gm);
    int first = __ffsll((long long)bm) - 1;
    if (lane == first && li >= 0) removed |= 1u << li;
    kth = gm;
  }

  const float scale = 0.125f;   // 1/sqrt(64)
  float ev[11]; float sum = 0.f;
#pragma unroll
  for (int i = 0; i < 11; ++i) {
    int idx = i * 32 + lane;
    float s = vals[i];
    float e = (idx < L_ && s >= kth) ? __expf((s - m0) * scale) : 0.f;
    ev[i] = e; sum += e;
  }
  sum = wave_sum32(sum);
  float inv = 1.f / sum;
#pragma unroll
  for (int i = 0; i < 11; ++i) {
    int idx = i * 32 + lane;
    if (idx < L_) p[idx] = ev[i] * inv;
  }
}

// ---------------------------------------------------------------------------
// gattn context: ctx[b,i,h,e] = sum_j A[b,h,i,j]*v[b,j,h,e]  (K=321, zero-pad)
// ---------------------------------------------------------------------------
__global__ __launch_bounds__(256)
void gattn_av(const float* __restrict__ Amat, const float* __restrict__ v,
              float* __restrict__ ctx)
{
  __shared__ __bf16 As[128 * LDSPAD];  // [i][j]
  __shared__ __bf16 Vs[64 * LDSPAD];   // [e][j]

  const int tid  = threadIdx.x;
  const int lane = tid & 31;
  const int wave = tid >> 5;
  const int mBlk = blockIdx.x * 128;
  const int bh   = blockIdx.z;
  const int b    = bh >> 3, h = bh & 7;
  const float* Ab = Amat + (size_t)bh * L_ * L_;
  const float* vb = v + (size_t)b * L_ * D_ + h * 64;
  const int mW = wave * 16;

  v8f acc[4] = {};

  for (int k0 = 0; k0 < L_; k0 += 32) {
#pragma unroll
    for (int t = 0; t < 4; ++t) {        // A tile 128x32 (rows unaligned)
      int idx4 = t * 256 + tid;
      int r = idx4 >> 3, c4 = (idx4 & 7) << 2;
      int gr = mBlk + r;
      int grc = gr < L_ ? gr : L_ - 1;
      float4 f;
      float* fp = (float*)&f;
#pragma unroll
      for (int j = 0; j < 4; ++j) {
        int gc = k0 + c4 + j;
        int gcc = gc < L_ ? gc : L_ - 1;
        float x = Ab[(size_t)grc * L_ + gcc];
        fp[j] = (gr < L_ && gc < L_) ? x : 0.f;
      }
      *(v4bf*)(As + r * LDSPAD + c4) = cvt4<ACT_NONE>(f);
    }
#pragma unroll
    for (int t = 0; t < 2; ++t) {        // V tile 32(j)x64(e) -> [e][j]
      int idx4 = t * 256 + tid;
      int kk = idx4 >> 4, e4 = (idx4 & 15) << 2;
      int gj = k0 + kk;
      int gjc = gj < L_ ? gj : L_ - 1;
      float4 f = *(const float4*)(vb + (size_t)gjc * D_ + e4);
      bool ok = gj < L_;
      f.x = ok ? f.x : 0.f; f.y = ok ? f.y : 0.f;
      f.z = ok ? f.z : 0.f; f.w = ok ? f.w : 0.f;
      v4bf h = cvt4<ACT_NONE>(f);
      Vs[(e4 + 0) * LDSPAD + kk] = h[0];
      Vs[(e4 + 1) * LDSPAD + kk] = h[1];
      Vs[(e4 + 2) * LDSPAD + kk] = h[2];
      Vs[(e4 + 3) * LDSPAD + kk] = h[3];
    }
    __syncthreads();
    v16bf af = frag_a(As, mW + (lane & 15), lane);
#pragma unroll
    for (int t = 0; t < 4; ++t) {
      v16bf bf = frag_b(Vs, t * 16 + (lane & 15), lane);
      acc[t] = __builtin_amdgcn_wmma_f32_16x16x32_bf16(
          false, af, false, bf, (short)0, acc[t], false, false);
    }
    __syncthreads();
  }

  const int rowBase = mBlk + mW + ((lane & 16) ? 8 : 0);
  const bool fullT = (mBlk + 128 <= L_);
  if (fullT) {
#pragma unroll
    for (int t = 0; t < 4; ++t) {
      const int col = t * 16 + (lane & 15);      // e < 64
      size_t base = ((size_t)b * L_ + rowBase) * D_ + h * 64 + col;
#pragma unroll
      for (int r = 0; r < 8; ++r, base += D_) ctx[base] = acc[t][r];
    }
  } else {
#pragma unroll
    for (int t = 0; t < 4; ++t) {
      const int col = t * 16 + (lane & 15);
      size_t base = ((size_t)b * L_ + rowBase) * D_ + h * 64 + col;
#pragma unroll
      for (int r = 0; r < 8; ++r, base += D_)
        if (rowBase + r < L_) ctx[base] = acc[t][r];
    }
  }
}

// ---------------------------------------------------------------------------
// Pointwise attention core (per token): V = softmax(q k^T / 4) @ v.
// Flash-style online softmax; one thread per query row; blockDim = pred/16.
// ---------------------------------------------------------------------------
__global__ void pwattn_core(const float* __restrict__ q,
                            const float* __restrict__ k,
                            const float* __restrict__ v,
                            float* __restrict__ out, int pred, int S)
{
  extern __shared__ float sm[];
  float* qs = sm;
  float* ks = sm + pred;
  float* vs = ks + S;

  const int n = blockIdx.x;
  const int tid = threadIdx.x;
  const int Sc = S >> 4;

  const float4* q4 = (const float4*)(q + (size_t)n * pred);
  const float4* k4 = (const float4*)(k + (size_t)n * S);
  const float4* v4 = (const float4*)(v + (size_t)n * S);
  for (int i = tid; i < (pred >> 2); i += blockDim.x) ((float4*)qs)[i] = q4[i];
  for (int i = tid; i < (S >> 2); i += blockDim.x) {
    ((float4*)ks)[i] = k4[i];
    ((float4*)vs)[i] = v4[i];
  }
  __syncthreads();

  const int l = tid;
  const float scale = 0.25f;     // 1/sqrt(16)
  float m = -INFINITY, d = 0.f, acc[16] = {};
  for (int s = 0; s < Sc; ++s) {
    float dot = 0.f;
#pragma unroll
    for (int e = 0; e < 16; ++e) dot += qs[l * 16 + e] * ks[s * 16 + e];
    float t  = dot * scale;
    float mn = fmaxf(m, t);
    float corr = __expf(m - mn);
    float w    = __expf(t - mn);
    d = d * corr + w;
#pragma unroll
    for (int e = 0; e < 16; ++e) acc[e] = acc[e] * corr + w * vs[s * 16 + e];
    m = mn;
  }
  float invd = 1.f / d;
  float* orow = out + (size_t)n * pred;
#pragma unroll
  for (int e = 0; e < 16; ++e) orow[l * 16 + e] = acc[e] * invd;
}

// ---------------------------------------------------------------------------
// z = LayerNorm(elu(x)) * g + b over DFF=2048, one block (256 thr) per row.
// ---------------------------------------------------------------------------
__global__ __launch_bounds__(256)
void ln_elu(const float* __restrict__ x, const float* __restrict__ g,
            const float* __restrict__ b, float* __restrict__ out)
{
  __shared__ float red[2][8];
  const int n = blockIdx.x, tid = threadIdx.x;
  const float* xr = x + (size_t)n * DFF_;

  float loc[8], s = 0.f, s2 = 0.f;
#pragma unroll
  for (int i = 0; i < 8; ++i) {
    float xv = xr[tid + i * 256];
    float e = xv > 0.f ? xv : __expf(xv) - 1.f;   // elu
    loc[i] = e; s += e; s2 += e * e;
  }
  s = wave_sum32(s); s2 = wave_sum32(s2);
  const int wave = tid >> 5, lane = tid & 31;
  if (lane == 0) { red[0][wave] = s; red[1][wave] = s2; }
  __syncthreads();
  float ts = 0.f, ts2 = 0.f;
#pragma unroll
  for (int w = 0; w < 8; ++w) { ts += red[0][w]; ts2 += red[1][w]; }
  const float mean = ts * (1.f / DFF_);
  const float var  = ts2 * (1.f / DFF_) - mean * mean;
  const float inv  = rsqrtf(var + 1e-5f);
  float* orow = out + (size_t)n * DFF_;
#pragma unroll
  for (int i = 0; i < 8; ++i) {
    int col = tid + i * 256;
    orow[col] = (loc[i] - mean) * inv * g[col] + b[col];
  }
}

// ---------------------------------------------------------------------------
// Host orchestration
// ---------------------------------------------------------------------------
extern "C" void kernel_launch(void* const* d_in, const int* in_sizes, int n_in,
                              void* d_out, int out_size, void* d_ws, size_t ws_size,
                              hipStream_t stream)
{
  const float* x    = (const float*)d_in[0];
  const float* aq_w = (const float*)d_in[1];  const float* aq_b = (const float*)d_in[2];
  const float* ak_w = (const float*)d_in[3];  const float* ak_b = (const float*)d_in[4];
  const float* av_w = (const float*)d_in[5];  const float* av_b = (const float*)d_in[6];
  const float* ao_w = (const float*)d_in[7];  const float* ao_b = (const float*)d_in[8];
  const float* en_w = (const float*)d_in[9];  const float* en_b = (const float*)d_in[10];
  const float* f1q_w = (const float*)d_in[11]; const float* f1q_b = (const float*)d_in[12];
  const float* f1k_w = (const float*)d_in[13]; const float* f1k_b = (const float*)d_in[14];
  const float* f1v_w = (const float*)d_in[15]; const float* f1v_b = (const float*)d_in[16];
  const float* f1o_w = (const float*)d_in[17]; const float* f1o_b = (const float*)d_in[18];
  const float* f2q_w = (const float*)d_in[19]; const float* f2q_b = (const float*)d_in[20];
  const float* f2k_w = (const float*)d_in[21]; const float* f2k_b = (const float*)d_in[22];
  const float* f2v_w = (const float*)d_in[23]; const float* f2v_b = (const float*)d_in[24];
  const float* f2o_w = (const float*)d_in[25]; const float* f2o_b = (const float*)d_in[26];
  const float* ln_g  = (const float*)d_in[27]; const float* ln_b  = (const float*)d_in[28];
  (void)in_sizes; (void)n_in; (void)out_size; (void)ws_size;

  const size_t S512  = (size_t)NTOK * D_;
  const size_t S2048 = (size_t)NTOK * DFF_;

  float* ws = (float*)d_ws;
  float* r0 = ws;              // q   -> y2
  float* r1 = ws + 1 * S512;   // k   -> k1 -> query2
  float* r2 = ws + 2 * S512;   // v   -> v1 -> V2
  float* r3 = ws + 3 * S512;   // ctx
  float* r4 = ws + 4 * S512;   // y1
  float* g0 = ws + 5 * S512;   // query1 -> z
  float* g1 = g0 + S2048;      // V1     -> k2
  float* g2 = g1 + S2048;      // t1     -> v2

  float* yOut = (float*)d_out;          // [B, L, D]
  float* Aout = yOut + S512;            // [B, H, L, L]

  const dim3 blk(256);
  auto gg = [](int M, int N) { return dim3((unsigned)((M + 127) / 128), (unsigned)(N / 64)); };

  // ---- gattn projections
  gemm_ba<ACT_RELU, ACT_NONE, false><<<gg(NTOK, D_), blk, 0, stream>>>(x, aq_w, aq_b, nullptr, r0, NTOK, D_, D_);
  gemm_ba<ACT_NONE, ACT_NONE, false><<<gg(NTOK, D_), blk, 0, stream>>>(x, ak_w, ak_b, nullptr, r1, NTOK, D_, D_);
  gemm_ba<ACT_RELU, ACT_NONE, false><<<gg(NTOK, D_), blk, 0, stream>>>(x, av_w, av_b, nullptr, r2, NTOK, D_, D_);

  // ---- scores -> top-k softmax (in-place in d_out A region) -> A@V
  gattn_scores<<<dim3(3, 6, B_ * H_), blk, 0, stream>>>(r0, r1, Aout);
  topk_softmax<<<dim3((B_ * H_ * L_) / 8), blk, 0, stream>>>(Aout);
  gattn_av<<<dim3(3, 1, B_ * H_), blk, 0, stream>>>(Aout, r2, r3);

  // ---- output projection + residual, encoder (leaky pre-activation)
  gemm_ba<ACT_NONE, ACT_NONE, true ><<<gg(NTOK, D_), blk, 0, stream>>>(r3, ao_w, ao_b, x, r4, NTOK, D_, D_);
  gemm_ba<ACT_NONE, ACT_LEAKY, false><<<gg(NTOK, D_), blk, 0, stream>>>(r4, en_w, en_b, nullptr, r0, NTOK, D_, D_);

  // ---- ff1 pointwise attention + LN(elu(.))
  gemm_ba<ACT_RELU, ACT_NONE, false><<<gg(NTOK, DFF_), blk, 0, stream>>>(r0, f1q_w, f1q_b, nullptr, g0, NTOK, D_, DFF_);
  gemm_ba<ACT_NONE, ACT_NONE, false><<<gg(NTOK, D_),   blk, 0, stream>>>(r0, f1k_w, f1k_b, nullptr, r1, NTOK, D_, D_);
  gemm_ba<ACT_RELU, ACT_NONE, false><<<gg(NTOK, D_),   blk, 0, stream>>>(r0, f1v_w, f1v_b, nullptr, r2, NTOK, D_, D_);
  pwattn_core<<<dim3(NTOK), dim3(DFF_ / 16), (DFF_ + 2 * D_) * sizeof(float), stream>>>(g0, r1, r2, g1, DFF_, D_);
  gemm_ba<ACT_NONE, ACT_NONE, true ><<<gg(NTOK, DFF_), blk, 0, stream>>>(g1, f1o_w, f1o_b, g0, g2, NTOK, DFF_, DFF_);
  ln_elu<<<dim3(NTOK), blk, 0, stream>>>(g2, ln_g, ln_b, g0);

  // ---- ff2 pointwise attention
  gemm_ba<ACT_RELU, ACT_NONE, false><<<gg(NTOK, D_),   blk, 0, stream>>>(g0, f2q_w, f2q_b, nullptr, r1, NTOK, DFF_, D_);
  gemm_ba<ACT_NONE, ACT_NONE, false><<<gg(NTOK, DFF_), blk, 0, stream>>>(g0, f2k_w, f2k_b, nullptr, g1, NTOK, DFF_, DFF_);
  gemm_ba<ACT_RELU, ACT_NONE, false><<<gg(NTOK, DFF_), blk, 0, stream>>>(g0, f2v_w, f2v_b, nullptr, g2, NTOK, DFF_, DFF_);
  pwattn_core<<<dim3(NTOK), dim3(D_ / 16), (D_ + 2 * DFF_) * sizeof(float), stream>>>(r1, g1, g2, r2, D_, DFF_);
  gemm_ba<ACT_NONE, ACT_NONE, true ><<<gg(NTOK, D_), blk, 0, stream>>>(r2, f2o_w, f2o_b, r1, yOut, NTOK, D_, D_);
}
